// DeepSpeedMoEMLP_50285477101612
// MI455X (gfx1250) — matrix operand order repaired and checked by reading, not compiled, not run on hardware
//
#include <hip/hip_runtime.h>
#include <cstdint>

// ---------------------------------------------------------------------------
// DeepSpeed MoE expert MLP:  out = gelu_tanh(x @ W1 + b1) @ W2 + b2
// M = B*S = 4096, H = 4096, F = 16384, fp32 in/out.
//
// Compute-bound (~1.6 kFLOP/byte at 23.3 TB/s HBM) -> bf16 WMMA
// (v_wmma_f32_16x16x32_bf16) with fp32 accumulation; intermediate kept as
// bf16 in d_ws.
//
// fp32->bf16 conversion: round-half-up by adding 0x8000 to the float bits
// (1 VALU), leaving the bf16 value in the high 16 bits.  Pairs packed with
// a single v_perm_b32; scalar stores fold into ds/global *_d16_hi stores.
//
// Per block: 128x128 C tile, 256 threads = 8 wave32 waves (4x2 wave grid),
// each wave owns 32x64 = 2x4 WMMA accumulators. K staged in 32-wide bf16
// LDS tiles, double-buffered: global loads for tile k+1 are issued before
// the WMMAs of tile k, staged in registers, and committed to the other LDS
// buffer with a single barrier per iteration.
// ---------------------------------------------------------------------------

typedef __attribute__((ext_vector_type(16))) __bf16 v16bf;
typedef __attribute__((ext_vector_type(8)))  float  v8f;

union FragU { v16bf v; uint4 q[2]; };

// float bits + 0x8000: bf16 (round-half-up) sits in the high 16 bits.
__device__ __forceinline__ unsigned int bfbits(float f) {
  union { float f; unsigned int u; } c; c.f = f;
  return c.u + 0x8000u;
}

// pack two f32 -> two bf16 in one dword: 2 adds + 1 v_perm_b32.
__device__ __forceinline__ unsigned int pk2bf(float a, float b) {
  unsigned int ua = bfbits(a);            // bf16(a) in [31:16]
  unsigned int ub = bfbits(b);            // bf16(b) in [31:16]
  // result bytes: [1:0] = ua[3:2], [3:2] = ub[3:2]
  return __builtin_amdgcn_perm(ub, ua, 0x07060302u);
}

__device__ __forceinline__ float gelu_tanh(float x) {
  const float k0 = 0.7978845608028654f;   // sqrt(2/pi)
  const float k1 = 0.044715f;
  float y = k0 * (x + k1 * x * x * x);
#if __has_builtin(__builtin_amdgcn_tanhf)
  float t = __builtin_amdgcn_tanhf(y);    // native v_tanh_f32 on gfx1250
#else
  float t = 1.0f - 2.0f / (1.0f + __expf(2.0f * y));
#endif
  return 0.5f * x * (1.0f + t);
}

// ---------------------------------------------------------------------------
// GEMM1: Out_bf16[M,N] = gelu(A_f32[M,K] @ B_f32[K,N] + bias[N])
// ---------------------------------------------------------------------------
__global__ __launch_bounds__(256) void mlp_gemm1_gelu_bf16(
    const float* __restrict__ A, const float* __restrict__ B,
    const float* __restrict__ bias, unsigned short* __restrict__ Out,
    int Kdim, int Ndim) {
  __shared__ unsigned short As[2][128 * 32];   // [row(128)][k(32)]
  __shared__ unsigned short Bs[2][128 * 32];   // transposed: [n(128)][k(32)]

  const int tid  = threadIdx.x;
  const int wave = tid >> 5;
  const int lane = tid & 31;
  const int l16  = lane & 15;
  const int lh   = (lane >> 4) & 1;
  const int mBase = blockIdx.y * 128;
  const int nBase = blockIdx.x * 128;
  const int wM = (wave >> 1) * 32;   // 0,32,64,96
  const int wN = (wave & 1) * 64;    // 0,64

  v8f acc[2][4];
  #pragma unroll
  for (int mt = 0; mt < 2; ++mt)
    #pragma unroll
    for (int nt = 0; nt < 4; ++nt)
      acc[mt][nt] = (v8f){0.f, 0.f, 0.f, 0.f, 0.f, 0.f, 0.f, 0.f};

  float4 ra[4], rb[4];

  auto load_global = [&](int kt) {
    #pragma unroll
    for (int p = 0; p < 4; ++p) {
      int g = tid + p * 256;
      int arow = g >> 3, acol = (g & 7) << 2;     // A: 8 float4 groups / row
      ra[p] = *(const float4*)(A + (size_t)(mBase + arow) * Kdim + kt + acol);
      int brow = g >> 5, bcol = (g & 31) << 2;    // B: 32 float4 groups / row
      rb[p] = *(const float4*)(B + (size_t)(kt + brow) * Ndim + nBase + bcol);
    }
  };

  auto store_lds = [&](int buf) {
    #pragma unroll
    for (int p = 0; p < 4; ++p) {
      int g = tid + p * 256;
      int arow = g >> 3, acol = (g & 7) << 2;
      *(uint2*)&As[buf][arow * 32 + acol] =
          make_uint2(pk2bf(ra[p].x, ra[p].y), pk2bf(ra[p].z, ra[p].w));
      int brow = g >> 5, bcol = (g & 31) << 2;
      // high-half stores fold into ds_store_b16_d16_hi (1 VALU add each)
      Bs[buf][(bcol + 0) * 32 + brow] = (unsigned short)(bfbits(rb[p].x) >> 16);
      Bs[buf][(bcol + 1) * 32 + brow] = (unsigned short)(bfbits(rb[p].y) >> 16);
      Bs[buf][(bcol + 2) * 32 + brow] = (unsigned short)(bfbits(rb[p].z) >> 16);
      Bs[buf][(bcol + 3) * 32 + brow] = (unsigned short)(bfbits(rb[p].w) >> 16);
    }
  };

  auto compute = [&](int buf) {
    // A 16x32 frag: lane l16 = row M; lanes 0-15 hold K {0..7,16..23},
    //               lanes 16-31 hold K {8..15,24..31}.
    v16bf af[2];
    #pragma unroll
    for (int mt = 0; mt < 2; ++mt) {
      int r = wM + mt * 16 + l16;
      FragU u;
      u.q[0] = *(const uint4*)&As[buf][r * 32 + lh * 8];
      u.q[1] = *(const uint4*)&As[buf][r * 32 + lh * 8 + 16];
      af[mt] = u.v;
    }
    // B 32x16 frag: lane l16 = col N; lanes 0-15 K {0..7,16..23}, lanes 16-31 K {8..15,24..31}.
    v16bf bfr[4];
    #pragma unroll
    for (int nt = 0; nt < 4; ++nt) {
      int c = wN + nt * 16 + l16;
      FragU u;
      u.q[0] = *(const uint4*)&Bs[buf][c * 32 + lh * 8];
      u.q[1] = *(const uint4*)&Bs[buf][c * 32 + lh * 8 + 16];
      bfr[nt] = u.v;
    }
    #pragma unroll
    for (int mt = 0; mt < 2; ++mt)
      #pragma unroll
      for (int nt = 0; nt < 4; ++nt)
        acc[mt][nt] = __builtin_amdgcn_wmma_f32_16x16x32_bf16(
            false, af[mt], false, bfr[nt], (short)0, acc[mt][nt], false, false);
  };

  const int nk = Kdim >> 5;
  load_global(0);
  store_lds(0);
  __syncthreads();
  for (int kt = 0; kt < nk; ++kt) {
    if (kt + 1 < nk) load_global((kt + 1) << 5);
    compute(kt & 1);
    if (kt + 1 < nk) store_lds((kt + 1) & 1);
    __syncthreads();
  }

  // epilogue: bias + gelu, store bf16.  C/D: VGPR r -> M=r (lanes 0-15) or
  // M=r+8 (lanes 16-31), N = l16.
  #pragma unroll
  for (int mt = 0; mt < 2; ++mt)
    #pragma unroll
    for (int nt = 0; nt < 4; ++nt) {
      int colg = nBase + wN + nt * 16 + l16;
      float bv = bias[colg];
      #pragma unroll
      for (int r = 0; r < 8; ++r) {
        int rowg = mBase + wM + mt * 16 + r + lh * 8;
        Out[(size_t)rowg * Ndim + colg] =
            (unsigned short)(bfbits(gelu_tanh(acc[mt][nt][r] + bv)) >> 16);
      }
    }
}

// ---------------------------------------------------------------------------
// GEMM2: Out_f32[M,N] = A_bf16[M,K] @ B_f32[K,N] + bias[N]
// ---------------------------------------------------------------------------
__global__ __launch_bounds__(256) void mlp_gemm2_bias_f32(
    const unsigned short* __restrict__ A, const float* __restrict__ B,
    const float* __restrict__ bias, float* __restrict__ Out,
    int Kdim, int Ndim) {
  __shared__ unsigned short As[2][128 * 32];
  __shared__ unsigned short Bs[2][128 * 32];

  const int tid  = threadIdx.x;
  const int wave = tid >> 5;
  const int lane = tid & 31;
  const int l16  = lane & 15;
  const int lh   = (lane >> 4) & 1;
  const int mBase = blockIdx.y * 128;
  const int nBase = blockIdx.x * 128;
  const int wM = (wave >> 1) * 32;
  const int wN = (wave & 1) * 64;

  v8f acc[2][4];
  #pragma unroll
  for (int mt = 0; mt < 2; ++mt)
    #pragma unroll
    for (int nt = 0; nt < 4; ++nt)
      acc[mt][nt] = (v8f){0.f, 0.f, 0.f, 0.f, 0.f, 0.f, 0.f, 0.f};

  uint4  ra[2];     // A is already bf16: 2 x 16B per thread
  float4 rb[4];

  auto load_global = [&](int kt) {
    #pragma unroll
    for (int p = 0; p < 2; ++p) {
      int g = tid + p * 256;
      int arow = g >> 2, acol = (g & 3) << 3;   // 4 groups of 8 ushorts / row
      ra[p] = *(const uint4*)(A + (size_t)(mBase + arow) * Kdim + kt + acol);
    }
    #pragma unroll
    for (int p = 0; p < 4; ++p) {
      int g = tid + p * 256;
      int brow = g >> 5, bcol = (g & 31) << 2;
      rb[p] = *(const float4*)(B + (size_t)(kt + brow) * Ndim + nBase + bcol);
    }
  };

  auto store_lds = [&](int buf) {
    #pragma unroll
    for (int p = 0; p < 2; ++p) {
      int g = tid + p * 256;
      int arow = g >> 2, acol = (g & 3) << 3;
      *(uint4*)&As[buf][arow * 32 + acol] = ra[p];
    }
    #pragma unroll
    for (int p = 0; p < 4; ++p) {
      int g = tid + p * 256;
      int brow = g >> 5, bcol = (g & 31) << 2;
      Bs[buf][(bcol + 0) * 32 + brow] = (unsigned short)(bfbits(rb[p].x) >> 16);
      Bs[buf][(bcol + 1) * 32 + brow] = (unsigned short)(bfbits(rb[p].y) >> 16);
      Bs[buf][(bcol + 2) * 32 + brow] = (unsigned short)(bfbits(rb[p].z) >> 16);
      Bs[buf][(bcol + 3) * 32 + brow] = (unsigned short)(bfbits(rb[p].w) >> 16);
    }
  };

  auto compute = [&](int buf) {
    v16bf af[2];
    #pragma unroll
    for (int mt = 0; mt < 2; ++mt) {
      int r = wM + mt * 16 + l16;
      FragU u;
      u.q[0] = *(const uint4*)&As[buf][r * 32 + lh * 8];
      u.q[1] = *(const uint4*)&As[buf][r * 32 + lh * 8 + 16];
      af[mt] = u.v;
    }
    v16bf bfr[4];
    #pragma unroll
    for (int nt = 0; nt < 4; ++nt) {
      int c = wN + nt * 16 + l16;
      FragU u;
      u.q[0] = *(const uint4*)&Bs[buf][c * 32 + lh * 8];
      u.q[1] = *(const uint4*)&Bs[buf][c * 32 + lh * 8 + 16];
      bfr[nt] = u.v;
    }
    #pragma unroll
    for (int mt = 0; mt < 2; ++mt)
      #pragma unroll
      for (int nt = 0; nt < 4; ++nt)
        acc[mt][nt] = __builtin_amdgcn_wmma_f32_16x16x32_bf16(
            false, af[mt], false, bfr[nt], (short)0, acc[mt][nt], false, false);
  };

  const int nk = Kdim >> 5;
  load_global(0);
  store_lds(0);
  __syncthreads();
  for (int kt = 0; kt < nk; ++kt) {
    if (kt + 1 < nk) load_global((kt + 1) << 5);
    compute(kt & 1);
    if (kt + 1 < nk) store_lds((kt + 1) & 1);
    __syncthreads();
  }

  #pragma unroll
  for (int mt = 0; mt < 2; ++mt)
    #pragma unroll
    for (int nt = 0; nt < 4; ++nt) {
      int colg = nBase + wN + nt * 16 + l16;
      float bv = bias[colg];
      #pragma unroll
      for (int r = 0; r < 8; ++r) {
        int rowg = mBase + wM + mt * 16 + r + lh * 8;
        Out[(size_t)rowg * Ndim + colg] = acc[mt][nt][r] + bv;
      }
    }
}

extern "C" void kernel_launch(void* const* d_in, const int* in_sizes, int n_in,
                              void* d_out, int out_size, void* d_ws, size_t ws_size,
                              hipStream_t stream) {
  const float* x  = (const float*)d_in[0];   // [B,S,H]
  const float* w1 = (const float*)d_in[1];   // [H,F]
  const float* b1 = (const float*)d_in[2];   // [F]
  const float* w2 = (const float*)d_in[3];   // [F,H]
  const float* b2 = (const float*)d_in[4];   // [H]
  float* out = (float*)d_out;

  const int F = in_sizes[2];                 // 16384
  const int H = in_sizes[4];                 // 4096
  const int M = in_sizes[0] / H;             // B*S = 4096

  unsigned short* h_bf16 = (unsigned short*)d_ws;  // [M,F] bf16 intermediate

  dim3 blk(256);
  mlp_gemm1_gelu_bf16<<<dim3(F / 128, M / 128), blk, 0, stream>>>(
      x, w1, b1, h_bf16, H, F);
  mlp_gemm2_bias_f32<<<dim3(H / 128, M / 128), blk, 0, stream>>>(
      h_bf16, w2, b2, out, F, H);
}
